// SelfAttention_82403242541160
// MI455X (gfx1250) — compile-verified, matrix operand order checked by reading, not run on hardware
//
#include <hip/hip_runtime.h>

// ---------------------------------------------------------------------------
// Self-attention with summed-over-queries output, CDNA5 (gfx1250) WMMA f16.
//   out[b,k] = sum_n c[n] * V[n,k],  c[n] = sum_q softmax_q(QK^T/16)[n]
// GEMMs use v_wmma_f32_16x16x32_f16. Attention K-tiles are staged into LDS
// (async global->LDS when available), shared by all 8 waves of the block.
// ---------------------------------------------------------------------------

typedef __attribute__((ext_vector_type(16))) _Float16 v16h;
typedef __attribute__((ext_vector_type(8)))  float    v8f;
typedef __attribute__((ext_vector_type(4)))  int      v4i;

#define B_ 32
#define N_ 2048
#define D_ 256

#define KT   32            // keys staged per tile
#define NKT  (N_ / KT)     // 64 tiles
#define KPAD 264           // padded row stride (halfs): 528B -> conflict-free b128

#if defined(__AMDGCN__) && __has_builtin(__builtin_amdgcn_global_load_async_to_lds_b128)
#define USE_ASYNC_LDS 1
#else
#define USE_ASYNC_LDS 0
#endif

#if USE_ASYNC_LDS
typedef __attribute__((address_space(1))) v4i gv4i;   // global b128 payload
typedef __attribute__((address_space(3))) v4i lv4i;   // LDS b128 payload
#endif

static __device__ __forceinline__ v8f wmma_f16(v16h a, v16h b, v8f c) {
  // (neg_a, A, neg_b, B, c_mod, C, reuse_a, reuse_b)
  return __builtin_amdgcn_wmma_f32_16x16x32_f16(false, a, false, b, (short)0, c,
                                                false, false);
}

// 16x32 f16 fragment in the CDNA5 16-bit A-operand lane layout from a
// row-major f16 matrix with row stride `ld` (halfs). Also the B-operand
// layout when rows of `base` correspond to output columns.
// lane<16: row=row0+lane, K = kbase+{0..7,16..23}; lane>=16: row=row0+lane-16,
// K = kbase+{8..15,24..31} -> two contiguous 16-byte runs per lane.
static __device__ __forceinline__ v16h load_frag(const _Float16* __restrict__ base,
                                                 int ld, int row0, int kbase,
                                                 int lane) {
  const int r  = row0 + (lane & 15);
  const int ho = (lane >> 4) << 3;  // 0 or 8
  const _Float16* p = base + (size_t)r * ld + kbase + ho;
  v16h f;
  uint4* d = reinterpret_cast<uint4*>(&f);
  d[0] = *reinterpret_cast<const uint4*>(p);
  d[1] = *reinterpret_cast<const uint4*>(p + 16);
  return f;
}

// Cooperative stage of one KT x 256 f16 tile into LDS via async global->LDS.
// 256 threads x 4 b128 each = 16KB. ASYNCcnt += 4 per wave.
static __device__ __forceinline__ void stage_async_tile(
    const _Float16* __restrict__ src, _Float16* dst, int t) {
#if USE_ASYNC_LDS
#pragma unroll
  for (int j = 0; j < 4; ++j) {
    const int idx = j * 256 + t;   // 0..1023 b128 chunks
    const int row = idx >> 5;      // 32 chunks per 256-half row
    const int c16 = idx & 31;
    const unsigned long long gaddr =
        (unsigned long long)(src + (size_t)row * D_ + c16 * 8);
    const unsigned laddr =
        (unsigned)(unsigned long long)(dst + row * KPAD + c16 * 8);  // LDS offset
    __builtin_amdgcn_global_load_async_to_lds_b128(
        (gv4i*)gaddr, (lv4i*)laddr, 0, 0);
  }
#else
  (void)src; (void)dst; (void)t;
#endif
}

static __device__ __forceinline__ void wait_async0() {
#if USE_ASYNC_LDS
#if __has_builtin(__builtin_amdgcn_s_wait_asynccnt)
  __builtin_amdgcn_s_wait_asynccnt(0);
#else
  asm volatile("s_wait_asynccnt 0x0" ::: "memory");
#endif
#endif
}

// ---------------------------------------------------------------------------
// Prep kernels
// ---------------------------------------------------------------------------
__global__ void __launch_bounds__(256) prep_x(const float* __restrict__ td,
                                              const float* __restrict__ ue,
                                              _Float16* __restrict__ x) {
  const size_t i = ((size_t)blockIdx.x * blockDim.x + threadIdx.x) * 4;
  const float4 a = *reinterpret_cast<const float4*>(td + i);
  const float4 b = *reinterpret_cast<const float4*>(ue + i);
  _Float16 o[4];
  o[0] = (_Float16)(a.x + b.x);
  o[1] = (_Float16)(a.y + b.y);
  o[2] = (_Float16)(a.z + b.z);
  o[3] = (_Float16)(a.w + b.w);
  *reinterpret_cast<uint2*>(x + i) = *reinterpret_cast<const uint2*>(o);
}

__global__ void __launch_bounds__(256) prep_w(const float* __restrict__ wq,
                                              const float* __restrict__ wk,
                                              const float* __restrict__ wv,
                                              _Float16* __restrict__ wt) {
  const int idx   = blockIdx.x * 256 + threadIdx.x;   // [0, 3*65536)
  const int which = idx >> 16;
  const int rem   = idx & 65535;
  const int kk    = rem >> 8;    // output column (row of transposed)
  const int d     = rem & 255;   // contraction dim
  const float* w  = (which == 0) ? wq : ((which == 1) ? wk : wv);
  const float  s  = (which == 0) ? 0.0625f : 1.0f;    // 1/sqrt(256) folded
  wt[idx] = (_Float16)(w[d * 256 + kk] * s);
}

__global__ void __launch_bounds__(256) zero_f32(float* __restrict__ p) {
  p[blockIdx.x * 256 + threadIdx.x] = 0.0f;
}

// ---------------------------------------------------------------------------
// QKV projection: [65536 x 256] x [256 x 256] for q,k,v (B = pre-transposed W).
// ---------------------------------------------------------------------------
__global__ void __launch_bounds__(256) qkv_gemm(const _Float16* __restrict__ x,
                                                const _Float16* __restrict__ wt,
                                                _Float16* __restrict__ qkv) {
  const int mt    = blockIdx.x;
  const int which = blockIdx.y;
  const int wave  = threadIdx.x >> 5;
  const int lane  = threadIdx.x & 31;
  const _Float16* w   = wt  + (size_t)which * D_ * D_;
  _Float16*       out = qkv + (size_t)which * (size_t)(B_ * N_) * D_;
  const int row0 = mt * 16;
  const int n0   = wave * 16;
  const int n1   = (wave + 8) * 16;

  v8f acc0 = {}, acc1 = {};
#pragma unroll
  for (int cc = 0; cc < 8; ++cc) {
    const v16h a  = load_frag(x, D_, row0, cc * 32, lane);
    const v16h b0 = load_frag(w, D_, n0,   cc * 32, lane);
    const v16h b1 = load_frag(w, D_, n1,   cc * 32, lane);
    acc0 = wmma_f16(a, b0, acc0);
    acc1 = wmma_f16(a, b1, acc1);
  }
  const int col = lane & 15;
  const int rh  = (lane >> 4) << 3;
#pragma unroll
  for (int r = 0; r < 8; ++r) {
    out[(size_t)(row0 + r + rh) * D_ + n0 + col] = (_Float16)acc0[r];
    out[(size_t)(row0 + r + rh) * D_ + n1 + col] = (_Float16)acc1[r];
  }
}

// ---------------------------------------------------------------------------
// Attention column sums. Block = 8 waves, 128 query rows. K tiles (32 keys)
// staged once per block into LDS (double-buffered async), B-fragments read
// from LDS with a padded stride (conflict-free ds_load_b128).
// ---------------------------------------------------------------------------
__global__ void __launch_bounds__(256) attn_colsum(const _Float16* __restrict__ q,
                                                   const _Float16* __restrict__ k,
                                                   float* __restrict__ c) {
  __shared__ __align__(16) _Float16 ksh[2][KT][KPAD];  // 33.0 KB
  __shared__ float csh[N_];                            //  8.0 KB
  for (int i = threadIdx.x; i < N_; i += 256) csh[i] = 0.0f;

  const int b     = blockIdx.y;
  const int wave  = threadIdx.x >> 5;
  const int lane  = threadIdx.x & 31;
  const int qrow0 = blockIdx.x * 128 + wave * 16;
  const _Float16* qb = q + (size_t)b * N_ * D_;
  const _Float16* kb = k + (size_t)b * N_ * D_;

  // Resident Q fragments: 16 rows x 256 dims (64 VGPRs/lane).
  v16h qa[8];
#pragma unroll
  for (int cc = 0; cc < 8; ++cc) qa[cc] = load_frag(qb, D_, qrow0, cc * 32, lane);

  float m[8], l[8];
#pragma unroll
  for (int r = 0; r < 8; ++r) { m[r] = -3.0e38f; l[r] = 0.0f; }

  // Process one staged tile: 2 x (16-key subtile) = 16 WMMAs per wave.
  auto process_tile = [&](const _Float16* kbuf, int nt, int pass) {
#pragma unroll
    for (int sub = 0; sub < 2; ++sub) {
      v8f s = {};
#pragma unroll
      for (int cc = 0; cc < 8; ++cc) {
        const v16h bk = load_frag(kbuf, KPAD, sub * 16, cc * 32, lane);
        s = wmma_f16(qa[cc], bk, s);
      }
      if (pass == 0) {
#pragma unroll
        for (int r8 = 0; r8 < 8; ++r8) {
          const float v = s[r8];
          float mx = v;
#pragma unroll
          for (int dd = 8; dd >= 1; dd >>= 1) mx = fmaxf(mx, __shfl_xor(mx, dd, 32));
          const float mn = fmaxf(m[r8], mx);
          float p = __expf(v - mn);
#pragma unroll
          for (int dd = 8; dd >= 1; dd >>= 1) p += __shfl_xor(p, dd, 32);
          l[r8] = l[r8] * __expf(m[r8] - mn) + p;
          m[r8] = mn;
        }
      } else {
        float cs = 0.0f;
#pragma unroll
        for (int r8 = 0; r8 < 8; ++r8) cs += __expf(s[r8] - m[r8]) * l[r8];
        cs += __shfl_xor(cs, 16, 32);   // combine row halves 0-7 / 8-15
        if (lane < 16) atomicAdd(&csh[nt * KT + sub * 16 + lane], cs);
      }
    }
  };

  // Pass 0: online (rowmax, rowsum-exp).  Pass 1: softmax column sums.
  for (int pass = 0; pass < 2; ++pass) {
#if USE_ASYNC_LDS
    __syncthreads();                                     // buffers free
    stage_async_tile(kb, &ksh[0][0][0], threadIdx.x);    // prologue: tile 0
    for (int nt = 0; nt < NKT; ++nt) {
      const int buf = nt & 1;
      wait_async0();                                     // tile nt landed
      __syncthreads();                                   // ...for all waves
      if (nt + 1 < NKT)                                  // overlap next stage
        stage_async_tile(kb + (size_t)(nt + 1) * KT * D_,
                         &ksh[buf ^ 1][0][0], threadIdx.x);
      process_tile(&ksh[buf][0][0], nt, pass);
    }
#else
    for (int nt = 0; nt < NKT; ++nt) {
      const _Float16* src = kb + (size_t)nt * KT * D_;
      uint4 r[4];
#pragma unroll
      for (int j = 0; j < 4; ++j) {
        const int idx = j * 256 + threadIdx.x;
        r[j] = *reinterpret_cast<const uint4*>(src + (size_t)(idx >> 5) * D_ +
                                               (idx & 31) * 8);
      }
      __syncthreads();                                   // prev compute done
#pragma unroll
      for (int j = 0; j < 4; ++j) {
        const int idx = j * 256 + threadIdx.x;
        *reinterpret_cast<uint4*>(&ksh[0][idx >> 5][(idx & 31) * 8]) = r[j];
      }
      __syncthreads();
      process_tile(&ksh[0][0][0], nt, pass);
    }
#endif
    if (pass == 0) {
#pragma unroll
      for (int r8 = 0; r8 < 8; ++r8) l[r8] = 1.0f / l[r8];
    }
  }

  __syncthreads();
  for (int i = threadIdx.x; i < N_; i += 256)
    atomicAdd(&c[(size_t)b * N_ + i], csh[i]);
}

// ---------------------------------------------------------------------------
// out[b,:] = c[b,:] @ V[b]
// ---------------------------------------------------------------------------
__global__ void __launch_bounds__(256) out_gemv(const float* __restrict__ c,
                                                const _Float16* __restrict__ v,
                                                float* __restrict__ out) {
  __shared__ float csh[N_];
  const int b = blockIdx.x;
  for (int i = threadIdx.x; i < N_; i += 256) csh[i] = c[(size_t)b * N_ + i];
  __syncthreads();
  const _Float16* vb = v + (size_t)b * N_ * D_;
  float acc = 0.0f;
  for (int n = 0; n < N_; ++n)
    acc += csh[n] * (float)vb[(size_t)n * D_ + threadIdx.x];
  out[b * D_ + threadIdx.x] = acc;
}

// ---------------------------------------------------------------------------
extern "C" void kernel_launch(void* const* d_in, const int* in_sizes, int n_in,
                              void* d_out, int out_size, void* d_ws, size_t ws_size,
                              hipStream_t stream) {
  const float* td = (const float*)d_in[0];
  const float* ue = (const float*)d_in[1];
  const float* wq = (const float*)d_in[2];
  const float* wk = (const float*)d_in[3];
  const float* wv = (const float*)d_in[4];
  float* out = (float*)d_out;

  // Workspace (~135 MB):
  //   x_f16 [65536,256] | wt_f16 [3,256,256] | qkv_f16 [3,65536,256] | c [32,2048]
  const size_t xElems = (size_t)B_ * N_ * D_;  // 16,777,216
  char* ws = (char*)d_ws;
  _Float16* x   = (_Float16*)ws;  ws += xElems * sizeof(_Float16);
  _Float16* wt  = (_Float16*)ws;  ws += (size_t)3 * D_ * D_ * sizeof(_Float16);
  _Float16* qkv = (_Float16*)ws;  ws += 3 * xElems * sizeof(_Float16);
  float*    c   = (float*)ws;
  _Float16* qf = qkv;
  _Float16* kf = qkv + xElems;
  _Float16* vf = qkv + 2 * xElems;

  prep_x<<<(int)(xElems / 1024), 256, 0, stream>>>(td, ue, x);
  prep_w<<<(3 * D_ * D_) / 256, 256, 0, stream>>>(wq, wk, wv, wt);
  zero_f32<<<(B_ * N_) / 256, 256, 0, stream>>>(c);
  qkv_gemm<<<dim3((B_ * N_) / 16, 3), 256, 0, stream>>>(x, wt, qkv);
  attn_colsum<<<dim3(N_ / 128, B_), 256, 0, stream>>>(qf, kf, c);
  out_gemv<<<B_, 256, 0, stream>>>(c, vf, out);
}
